// StandardDecoder_46737834115643
// MI455X (gfx1250) — matrix operand order repaired and checked by reading, not compiled, not run on hardware
//
#include <hip/hip_runtime.h>
#include <hip/hip_bf16.h>

// ---------------------------------------------------------------------------
// StandardDecoder for MI455X (gfx1250): persistent wave32 WMMA decoder.
// B=64,T=256,S=512,IN=512,H=512,P=256,C=512
// ---------------------------------------------------------------------------

typedef __attribute__((ext_vector_type(16))) _Float16 v16h;
typedef __attribute__((ext_vector_type(8)))  _Float16 v8h;
typedef __attribute__((ext_vector_type(8)))  float    v8f;

static constexpr int Bn = 64, Tn = 256, Sn = 512, INn = 512, Hn = 512, Pn = 256, Cn = 512;
static constexpr int G4 = 4 * Hn;          // 2048 gate outputs
static constexpr int KX = INn + Pn;        // 768  = [x_t | feed]
static constexpr int K0 = KX + Hn;         // 1280 = layer0 total K
static constexpr int K1 = Hn + Hn;         // 1024 = layer1 total K
static constexpr int HC = Hn + Cn;         // 1024 = [h1 | ectx] (proj K)

// LDS partition (bytes): xcat 16x768 f16, h0 16x512 f16, hcat 16x1024 f16,
// q/alpha 16x512 f16, score 16x512 f32  => 122880 B
static constexpr int SMEM_BYTES = 16*KX*2 + 16*Hn*2 + 16*HC*2 + 16*Sn*2 + 16*Sn*4;

__device__ __forceinline__ v16h mk16(v8h lo, v8h hi) {
  v16h r;
#pragma unroll
  for (int i = 0; i < 8; ++i) { r[i] = lo[i]; r[i + 8] = hi[i]; }
  return r;
}

// A-fragment (16x32 f16): lane l -> row (l&15); K = kb+{0..7}, kb+16+{0..7}, kb=(l>>4)*8
__device__ __forceinline__ v16h loadA(const _Float16* base, int stride, int kk, int lane) {
  const _Float16* p = base + (size_t)(lane & 15) * stride + kk + ((lane >> 4) << 3);
  return mk16(*(const v8h*)p, *(const v8h*)(p + 16));
}
// A-fragment with all 16 rows identical (row-replicated vector, e.g. alpha)
__device__ __forceinline__ v16h loadArep(const _Float16* vec, int kk, int lane) {
  const _Float16* p = vec + kk + ((lane >> 4) << 3);
  return mk16(*(const v8h*)p, *(const v8h*)(p + 16));
}
// B-fragment (32x16 f16) from W stored row-major per output col: lane l -> out col
// (l&15), K = kb+{0..7}, kb+16+{0..7}, kb=(l>>4)*8
__device__ __forceinline__ v16h loadB(const _Float16* base, int stride, int kk, int lane) {
  const _Float16* p = base + (size_t)(lane & 15) * stride + kk + ((lane >> 4) << 3);
  return mk16(*(const v8h*)p, *(const v8h*)(p + 16));
}
// B-fragment with all 16 output cols identical (replicated vector, e.g. q)
__device__ __forceinline__ v16h loadBrep(const _Float16* vec, int kk, int lane) {
  const _Float16* p = vec + kk + ((lane >> 4) << 3);
  return mk16(*(const v8h*)p, *(const v8h*)(p + 16));
}

__device__ __forceinline__ v8f wmma(v16h a, v16h b, v8f c) {
  return __builtin_amdgcn_wmma_f32_16x16x32_f16(false, a, false, b, (short)0, c, false, false);
}

__device__ __forceinline__ float sigm(float x) { return 1.0f / (1.0f + __expf(-x)); }

// ------------------------------- prep kernels -------------------------------

__global__ void k_cvt(const float* __restrict__ s, _Float16* __restrict__ d, int n) {
  for (int i = blockIdx.x * blockDim.x + threadIdx.x; i < n; i += gridDim.x * blockDim.x)
    d[i] = (_Float16)s[i];
}

__global__ void k_concat_cvt(const float* __restrict__ a, const float* __restrict__ b,
                             _Float16* __restrict__ d, int rows, int ka, int kb) {
  int kt = ka + kb, tot = rows * kt;
  for (int i = blockIdx.x * blockDim.x + threadIdx.x; i < tot; i += gridDim.x * blockDim.x) {
    int r = i / kt, k = i - r * kt;
    float v = (k < ka) ? a[(size_t)r * ka + k] : b[(size_t)r * kb + (k - ka)];
    d[i] = (_Float16)v;
  }
}

__global__ void k_transpose_cvt(const float* __restrict__ s, _Float16* __restrict__ d,
                                int rows, int cols, int batch) {
  long tot = (long)batch * rows * cols;
  long rc = (long)rows * cols;
  for (long i = (long)blockIdx.x * blockDim.x + threadIdx.x; i < tot;
       i += (long)gridDim.x * blockDim.x) {
    long b = i / rc, rem = i - b * rc;
    int r = (int)(rem / cols), c = (int)(rem - (long)r * cols);
    d[b * rc + (long)c * rows + r] = (_Float16)s[i];   // d[b][c][r] = s[b][r][c]
  }
}

__global__ void k_add2(const float* __restrict__ a, const float* __restrict__ b,
                       float* __restrict__ d, int n) {
  for (int i = blockIdx.x * blockDim.x + threadIdx.x; i < n; i += gridDim.x * blockDim.x)
    d[i] = a[i] + b[i];
}

// ------------------------------- main kernel --------------------------------

__global__ void __launch_bounds__(512, 1)
decoder_kernel(const _Float16* __restrict__ dec16,   // [B][T][IN]
               const _Float16* __restrict__ ctx16,   // [B][S][C]
               const _Float16* __restrict__ ctxT16,  // [B][C][S]
               const float*    __restrict__ mask,    // [B][S]
               const _Float16* __restrict__ W0,      // [2048][1280] = [Wih0|Whh0]
               const _Float16* __restrict__ W1,      // [2048][1024] = [Wih1|Whh1]
               const _Float16* __restrict__ WattT,   // [C][H] (Watt^T)
               const _Float16* __restrict__ Wp,      // [P][1024]
               const float*    __restrict__ b0,      // [2048]
               const float*    __restrict__ b1,      // [2048]
               const float*    __restrict__ bp,      // [256]
               float* __restrict__ dec_out,          // [B][T][P]
               float* __restrict__ att_out)          // [B][T][S]
{
  extern __shared__ char smem[];
  _Float16* xcat = (_Float16*)smem;          // [16][768]  cols 0..511 x_t, 512..767 feed
  _Float16* h0s  = xcat + 16 * KX;           // [16][512]
  _Float16* hcat = h0s + 16 * Hn;            // [16][1024] cols 0..511 h1, 512..1023 ectx
  _Float16* q16  = hcat + 16 * HC;           // [16][512]  q; reused as alpha16
  float*    scor = (float*)(q16 + 16 * Sn);  // [16][512]

  const int tid = threadIdx.x;
  const int wave = tid >> 5;                 // 0..15
  const int lane = tid & 31;
  const int bbase = blockIdx.x * 16;         // 16 batch rows per WG
  const int nn = lane & 15;                  // D-tile column within tile
  const int mhi = (lane >> 4) << 3;          // D-tile row offset (+0 or +8)

  // zero state (t==0): feed, h0, h1 = 0; c in wave-private registers
  for (int i = tid; i < 16 * KX; i += 512) xcat[i] = (_Float16)0.f;
  for (int i = tid; i < 16 * Hn; i += 512) h0s[i] = (_Float16)0.f;
  for (int i = tid; i < 16 * HC; i += 512) hcat[i] = (_Float16)0.f;
  v8f c0[2], c1[2];
#pragma unroll
  for (int g = 0; g < 2; ++g)
#pragma unroll
    for (int e = 0; e < 8; ++e) { c0[g][e] = 0.f; c1[g][e] = 0.f; }
  __syncthreads();

  for (int t = 0; t < Tn; ++t) {
    // ---- stage x_t (f16) into xcat[:,0:512] ----
    for (int i = tid; i < 16 * INn; i += 512) {
      int m = i >> 9, k = i & 511;
      xcat[m * KX + k] = dec16[(((size_t)(bbase + m)) * Tn + t) * INn + k];
    }
    __syncthreads();

    // ---- layer 0 LSTM: gates = [x|feed] @ Wih0^T + h0 @ Whh0^T ----
    float hn0[2][8];
#pragma unroll
    for (int grp = 0; grp < 2; ++grp) {
      const int n0h = (wave * 2 + grp) * 16;
      v8f acc[4] = {{}, {}, {}, {}};
      const _Float16* wb[4];
#pragma unroll
      for (int G = 0; G < 4; ++G) wb[G] = W0 + (size_t)(G * Hn + n0h) * K0;
      for (int kk = 0; kk < KX; kk += 32) {              // A = xcat
        v16h a = loadA(xcat, KX, kk, lane);
#pragma unroll
        for (int G = 0; G < 4; ++G)
          acc[G] = wmma(a, loadB(wb[G], K0, kk, lane), acc[G]);
      }
      for (int kk = 0; kk < Hn; kk += 32) {              // A = h0 (prev)
        v16h a = loadA(h0s, Hn, kk, lane);
#pragma unroll
        for (int G = 0; G < 4; ++G)
          acc[G] = wmma(a, loadB(wb[G], K0, KX + kk, lane), acc[G]);
      }
      const float bi = b0[0 * Hn + n0h + nn], bf = b0[1 * Hn + n0h + nn];
      const float bg = b0[2 * Hn + n0h + nn], bo = b0[3 * Hn + n0h + nn];
#pragma unroll
      for (int e = 0; e < 8; ++e) {
        float cn = sigm(acc[1][e] + bf) * c0[grp][e] +
                   sigm(acc[0][e] + bi) * tanhf(acc[2][e] + bg);
        c0[grp][e] = cn;
        hn0[grp][e] = sigm(acc[3][e] + bo) * tanhf(cn);
      }
    }
    __syncthreads();                                     // all waves done reading h0s
#pragma unroll
    for (int grp = 0; grp < 2; ++grp) {
      const int n0h = (wave * 2 + grp) * 16;
#pragma unroll
      for (int e = 0; e < 8; ++e)
        h0s[(e + mhi) * Hn + n0h + nn] = (_Float16)hn0[grp][e];
    }
    __syncthreads();

    // ---- layer 1 LSTM: gates = h0 @ Wih1^T + h1 @ Whh1^T ----
    float hn1[2][8];
#pragma unroll
    for (int grp = 0; grp < 2; ++grp) {
      const int n0h = (wave * 2 + grp) * 16;
      v8f acc[4] = {{}, {}, {}, {}};
      const _Float16* wb[4];
#pragma unroll
      for (int G = 0; G < 4; ++G) wb[G] = W1 + (size_t)(G * Hn + n0h) * K1;
      for (int kk = 0; kk < Hn; kk += 32) {              // A = h0 (new)
        v16h a = loadA(h0s, Hn, kk, lane);
#pragma unroll
        for (int G = 0; G < 4; ++G)
          acc[G] = wmma(a, loadB(wb[G], K1, kk, lane), acc[G]);
      }
      for (int kk = 0; kk < Hn; kk += 32) {              // A = h1 (prev, hcat[:,0:512])
        v16h a = loadA(hcat, HC, kk, lane);
#pragma unroll
        for (int G = 0; G < 4; ++G)
          acc[G] = wmma(a, loadB(wb[G], K1, Hn + kk, lane), acc[G]);
      }
      const float bi = b1[0 * Hn + n0h + nn], bf = b1[1 * Hn + n0h + nn];
      const float bg = b1[2 * Hn + n0h + nn], bo = b1[3 * Hn + n0h + nn];
#pragma unroll
      for (int e = 0; e < 8; ++e) {
        float cn = sigm(acc[1][e] + bf) * c1[grp][e] +
                   sigm(acc[0][e] + bi) * tanhf(acc[2][e] + bg);
        c1[grp][e] = cn;
        hn1[grp][e] = sigm(acc[3][e] + bo) * tanhf(cn);
      }
    }
    __syncthreads();
#pragma unroll
    for (int grp = 0; grp < 2; ++grp) {
      const int n0h = (wave * 2 + grp) * 16;
#pragma unroll
      for (int e = 0; e < 8; ++e)
        hcat[(e + mhi) * HC + n0h + nn] = (_Float16)hn1[grp][e];
    }
    __syncthreads();

    // ---- q = h1 @ Watt  (B tile = Watt^T row-major) ----
#pragma unroll
    for (int qt = 0; qt < 2; ++qt) {
      const int n0 = (wave * 2 + qt) * 16;
      v8f acc = {};
      for (int kk = 0; kk < Hn; kk += 32)
        acc = wmma(loadA(hcat, HC, kk, lane),
                   loadB(WattT + (size_t)n0 * Hn, Hn, kk, lane), acc);
#pragma unroll
      for (int e = 0; e < 8; ++e)
        q16[(e + mhi) * Sn + n0 + nn] = (_Float16)acc[e];
    }
    __syncthreads();

    // ---- scores: wave r owns batch row r.  A = ctx[s-tile], B = q replicated ----
    {
      const int r = wave;
      const size_t bg = (size_t)(bbase + r);
      const _Float16* ctxb = ctx16 + bg * Sn * Cn;
      const _Float16* qrow = q16 + r * Sn;
      for (int st = 0; st < Sn / 16; ++st) {
        const int s0 = st * 16;
        v8f acc = {};
        for (int kk = 0; kk < Cn; kk += 32)
          acc = wmma(loadA(ctxb + (size_t)s0 * Cn, Cn, kk, lane),
                     loadBrep(qrow, kk, lane), acc);
        if ((lane & 15) == 0) {                          // only col 0 is meaningful
#pragma unroll
          for (int e = 0; e < 8; ++e) {
            const int s = s0 + e + mhi;
            scor[r * Sn + s] = acc[e] + (mask[bg * Sn + s] - 1.0f) * 1e9f;
          }
        }
      }
    }
    __syncthreads();

    // ---- softmax (wave-local over 512 values) + write attention weights ----
    {
      const int r = wave;
      const size_t bg = (size_t)(bbase + r);
      float mx = -3.4e38f;
      for (int s = lane; s < Sn; s += 32) mx = fmaxf(mx, scor[r * Sn + s]);
#pragma unroll
      for (int off = 16; off; off >>= 1) mx = fmaxf(mx, __shfl_xor(mx, off, 32));
      float sum = 0.f;
      for (int s = lane; s < Sn; s += 32) {
        float e = __expf(scor[r * Sn + s] - mx);
        scor[r * Sn + s] = e;
        sum += e;
      }
#pragma unroll
      for (int off = 16; off; off >>= 1) sum += __shfl_xor(sum, off, 32);
      const float inv = 1.0f / sum;
      for (int s = lane; s < Sn; s += 32) {
        float a = scor[r * Sn + s] * inv;
        att_out[(bg * Tn + t) * Sn + s] = a;
        q16[r * Sn + s] = (_Float16)a;                   // alpha16 (q16 reused)
      }
    }
    __syncthreads();

    // ---- ectx = alpha @ ctx : A = alpha replicated rows, B = ctx^T tiles ----
    {
      const int r = wave;
      const size_t bg = (size_t)(bbase + r);
      const _Float16* ctxTb = ctxT16 + bg * Cn * Sn;     // [C][S]
      const _Float16* arow = q16 + r * Sn;
      for (int ct = 0; ct < Cn / 16; ++ct) {
        const int n0 = ct * 16;
        v8f acc = {};
        for (int kk = 0; kk < Sn; kk += 32)
          acc = wmma(loadArep(arow, kk, lane),
                     loadB(ctxTb + (size_t)n0 * Sn, Sn, kk, lane), acc);
        if (lane < 16) hcat[r * HC + Hn + n0 + lane] = (_Float16)acc[0];
      }
    }
    __syncthreads();

    // ---- projection: res = tanh([h1|ectx] @ Wproj^T + bproj); feed = res ----
    {
      const int n0 = wave * 16;                          // 16 tiles, 16 waves
      v8f acc = {};
      for (int kk = 0; kk < HC; kk += 32)
        acc = wmma(loadA(hcat, HC, kk, lane),
                   loadB(Wp + (size_t)n0 * HC, HC, kk, lane), acc);
      const float bb = bp[n0 + nn];
#pragma unroll
      for (int e = 0; e < 8; ++e) {
        const int m = e + mhi;
        const float res = tanhf(acc[e] + bb);
        dec_out[(((size_t)(bbase + m)) * Tn + t) * Pn + n0 + nn] = res;
        xcat[m * KX + INn + n0 + nn] = (_Float16)res;    // input feeding
      }
    }
    __syncthreads();
  }
}

// --------------------------------- launch -----------------------------------

extern "C" void kernel_launch(void* const* d_in, const int* in_sizes, int n_in,
                              void* d_out, int out_size, void* d_ws, size_t ws_size,
                              hipStream_t stream) {
  const float* dec_input = (const float*)d_in[0];
  const float* ctx   = (const float*)d_in[1];
  const float* mask  = (const float*)d_in[2];
  const float* Wih0  = (const float*)d_in[3];
  const float* Whh0  = (const float*)d_in[4];
  const float* bih0  = (const float*)d_in[5];
  const float* bhh0  = (const float*)d_in[6];
  const float* Wih1  = (const float*)d_in[7];
  const float* Whh1  = (const float*)d_in[8];
  const float* bih1  = (const float*)d_in[9];
  const float* bhh1  = (const float*)d_in[10];
  const float* Watt  = (const float*)d_in[11];
  const float* Wproj = (const float*)d_in[12];
  const float* bproj = (const float*)d_in[13];

  size_t off = 0;
  char* ws = (char*)d_ws;
  auto take = [&](size_t elems, size_t esz) -> void* {
    void* p = ws + off;
    off += (elems * esz + 255) & ~(size_t)255;
    return p;
  };
  _Float16* dec16  = (_Float16*)take((size_t)Bn * Tn * INn, 2);   // 8.4M
  _Float16* ctx16  = (_Float16*)take((size_t)Bn * Sn * Cn, 2);    // 16.8M
  _Float16* ctxT16 = (_Float16*)take((size_t)Bn * Cn * Sn, 2);    // 16.8M
  _Float16* W0c    = (_Float16*)take((size_t)G4 * K0, 2);
  _Float16* W1c    = (_Float16*)take((size_t)G4 * K1, 2);
  _Float16* WattT  = (_Float16*)take((size_t)Cn * Hn, 2);
  _Float16* Wp16   = (_Float16*)take((size_t)Pn * HC, 2);
  float*    b0c    = (float*)take(G4, 4);
  float*    b1c    = (float*)take(G4, 4);
  (void)ws_size; (void)n_in; (void)in_sizes; (void)out_size;

  // one-shot f32 -> f16 layout pass (all on `stream`, graph-capture safe)
  k_cvt<<<2048, 256, 0, stream>>>(dec_input, dec16, Bn * Tn * INn);
  k_cvt<<<4096, 256, 0, stream>>>(ctx, ctx16, Bn * Sn * Cn);
  k_transpose_cvt<<<4096, 256, 0, stream>>>(ctx, ctxT16, Sn, Cn, Bn);
  k_concat_cvt<<<2048, 256, 0, stream>>>(Wih0, Whh0, W0c, G4, KX, Hn);
  k_concat_cvt<<<2048, 256, 0, stream>>>(Wih1, Whh1, W1c, G4, Hn, Hn);
  k_transpose_cvt<<<256, 256, 0, stream>>>(Watt, WattT, Hn, Cn, 1);
  k_cvt<<<256, 256, 0, stream>>>(Wproj, Wp16, Pn * HC);
  k_add2<<<8, 256, 0, stream>>>(bih0, bhh0, b0c, G4);
  k_add2<<<8, 256, 0, stream>>>(bih1, bhh1, b1c, G4);

  float* dec_out = (float*)d_out;                         // [B][T][P]
  float* att_out = dec_out + (size_t)Bn * Tn * Pn;        // [B][T][S]

  hipFuncSetAttribute((const void*)decoder_kernel,
                      hipFuncAttributeMaxDynamicSharedMemorySize, SMEM_BYTES);
  decoder_kernel<<<dim3(Bn / 16), dim3(512), SMEM_BYTES, stream>>>(
      dec16, ctx16, ctxT16, mask, W0c, W1c, WattT, Wp16, b0c, b1c, bproj,
      dec_out, att_out);
}
